// NLMConv1DBlock_15857019257192
// MI455X (gfx1250) — compile-verified, matrix operand order checked by reading, not run on hardware
//
#include <hip/hip_runtime.h>
#include <hip/hip_bf16.h>

typedef __attribute__((ext_vector_type(16))) _Float16 v16h;
typedef __attribute__((ext_vector_type(8)))  float    v8f;

// Problem constants (from reference setup_inputs)
#define BB   256   // b*t
#define NN   64    // n
#define DD   32    // D
#define TT   16    // t
#define Bb   16    // b

// Fast sigmoid: v_exp_f32 + v_add + v_rcp_f32 (avoids IEEE div expansion)
__device__ __forceinline__ float sigf(float z) {
    return __builtin_amdgcn_rcpf(1.0f + __expf(-z));
}
__device__ __forceinline__ float swishf(float y) {
    return y * sigf(y);
}

// Compile-time K base for element e (0..15) of a 16x32 f16 fragment; the
// runtime half-wave offset (+8 for lanes 16-31) is applied separately so all
// control flow stays compile-time. Layout per CDNA5 ISA 7.12.2.
__device__ __forceinline__ constexpr int frag_kb(int e) {
    int v = e >> 1;
    return (v < 4 ? 2 * v : 16 + 2 * (v - 4)) + (e & 1);
}

__device__ __forceinline__ v8f wmma16(v16h a, v16h b, v8f c) {
    return __builtin_amdgcn_wmma_f32_16x16x32_f16(
        /*neg_a=*/false, a, /*neg_b=*/false, b,
        /*c_mod=*/(short)0, c, /*reuse_a=*/false, /*reuse_b=*/false);
}

// ---------------------------------------------------------------------------
// g0: per-batch max/min over n of the 2 input channels -> sigmoid(4->32)
// ---------------------------------------------------------------------------
__global__ __launch_bounds__(64) void k_g0(const float* __restrict__ x,
                                           const float* __restrict__ W1_0,
                                           const float* __restrict__ b1_0,
                                           float* __restrict__ g0) {
    int bb = blockIdx.x;
    int tid = threadIdx.x;
    __shared__ float sm[4][64];
    float x0 = x[(bb * NN + tid) * 2 + 0];
    float x1 = x[(bb * NN + tid) * 2 + 1];
    sm[0][tid] = x0; sm[1][tid] = x0; sm[2][tid] = x1; sm[3][tid] = x1;
    __syncthreads();
    for (int s = 32; s > 0; s >>= 1) {
        if (tid < s) {
            sm[0][tid] = fmaxf(sm[0][tid], sm[0][tid + s]);
            sm[1][tid] = fminf(sm[1][tid], sm[1][tid + s]);
            sm[2][tid] = fmaxf(sm[2][tid], sm[2][tid + s]);
            sm[3][tid] = fminf(sm[3][tid], sm[3][tid + s]);
        }
        __syncthreads();
    }
    if (tid < DD) {
        // reduce_unary layout: [max_c0, min_c0, max_c1, min_c1]
        float z = W1_0[tid * 4 + 0] * sm[0][0] + W1_0[tid * 4 + 1] * sm[1][0] +
                  W1_0[tid * 4 + 2] * sm[2][0] + W1_0[tid * 4 + 3] * sm[3][0] +
                  b1_0[tid];
        g0[bb * DD + tid] = sigf(z);
    }
}

// ---------------------------------------------------------------------------
// g1: per (b,i): features [x_i0, x_i1, e0, a0, e1, a1, e2, a2] -> sigmoid(8->32)
// ---------------------------------------------------------------------------
__global__ __launch_bounds__(256) void k_g1(const float* __restrict__ x,
                                            const float* __restrict__ W1_1,
                                            const float* __restrict__ b1_1,
                                            float* __restrict__ g1) {
    int idx = blockIdx.x * blockDim.x + threadIdx.x;
    if (idx >= BB * NN) return;
    int bb = idx >> 6, i = idx & 63;
    float xi0 = x[(bb * NN + i) * 2 + 0];
    float xi1 = x[(bb * NN + i) * 2 + 1];
    float e2 = 0.0f, a2 = 1.0f;
    for (int j = 0; j < NN; j++) {
        float c = (xi1 < x[(bb * NN + j) * 2 + 1]) ? 1.0f : 0.0f;
        bool diag = (j == i);
        e2 = fmaxf(e2, diag ? 0.0f : c);
        a2 = fminf(a2, diag ? 1.0f : c);
    }
    float f[8];
    f[0] = xi0; f[1] = xi1;
    f[2] = (i < NN - 1) ? 1.0f : 0.0f;  // e0: max of (i<j), diag excluded
    f[3] = (i == 0) ? 1.0f : 0.0f;      // a0: min of (i<j), diag excluded
    f[4] = 0.0f; f[5] = 0.0f;           // (i==j) off-diagonal
    f[6] = e2; f[7] = a2;
    for (int d = 0; d < DD; d++) {
        float z = b1_1[d];
        #pragma unroll
        for (int k = 0; k < 8; k++) z += W1_1[d * 8 + k] * f[k];
        g1[idx * DD + d] = sigf(z);
    }
}

// ---------------------------------------------------------------------------
// g2 + fused reduce_pair: one wave per (b,i). 64 j-rows x 10 features -> 32
// outputs via 2 WMMAs per 16-row j-tile (K padded 10->32). Sigmoid + diagonal
// mask + max/min over j fused into the C fragment; g2 (128 MB) never hits
// memory. All fragment fills are branchless (cndmask selects only).
// ---------------------------------------------------------------------------
__global__ __launch_bounds__(32) void k_g2red(const float* __restrict__ x,
                                              const float* __restrict__ W1_2,
                                              const float* __restrict__ b1_2,
                                              float* __restrict__ red) {
    int gid = blockIdx.x;          // 0 .. BB*NN-1
    int bb = gid >> 6;
    int i = gid & 63;
    int lane = threadIdx.x;        // 0..31
    int col = lane & 15;           // C/D column within a 16-wide N tile
    bool hh = (lane >> 4) != 0;    // half-wave select

    float xi0 = x[(bb * NN + i) * 2 + 0];
    float xi1 = x[(bb * NN + i) * 2 + 1];

    // B fragments: W1_2 is (32,10); pad K 10->32 decided at compile time.
    v16h bf0, bf1;
    #pragma unroll
    for (int e = 0; e < 16; e++) {
        const int kb = frag_kb(e);          // compile-time
        float lo0 = (kb < 10)     ? W1_2[col * 10 + kb]            : 0.0f;
        float hi0 = (kb + 8 < 10) ? W1_2[col * 10 + kb + 8]        : 0.0f;
        float lo1 = (kb < 10)     ? W1_2[(col + 16) * 10 + kb]     : 0.0f;
        float hi1 = (kb + 8 < 10) ? W1_2[(col + 16) * 10 + kb + 8] : 0.0f;
        bf0[e] = (_Float16)(hh ? hi0 : lo0);
        bf1[e] = (_Float16)(hh ? hi1 : lo1);
    }

    float bias0 = b1_2[col], bias1 = b1_2[col + 16];
    float emax0 = 0.0f, amin0 = 1.0f, emax1 = 0.0f, amin1 = 1.0f;

    #pragma unroll
    for (int jt = 0; jt < 4; jt++) {
        int j = jt * 16 + (lane & 15);           // A row M = lane&15
        float xj0 = x[(bb * NN + j) * 2 + 0];
        float xj1 = x[(bb * NN + j) * 2 + 1];
        // feature at literal K (switch over a literal -> constant-folded)
        auto fv = [&](int k) -> float {
            switch (k) {
                case 0: return xi0;
                case 1: return xi1;
                case 2: return (i < j) ? 1.0f : 0.0f;
                case 3: return (i == j) ? 1.0f : 0.0f;
                case 4: return (xi1 < xj1) ? 1.0f : 0.0f;
                case 5: return xj0;
                case 6: return xj1;
                case 7: return (j < i) ? 1.0f : 0.0f;
                case 8: return (j == i) ? 1.0f : 0.0f;
                case 9: return (xj1 < xi1) ? 1.0f : 0.0f;
                default: return 0.0f;
            }
        };
        v16h af;
        #pragma unroll
        for (int e = 0; e < 16; e++) {
            const int kb = frag_kb(e);           // compile-time
            float lo = fv(kb);
            float hi = fv(kb + 8);
            af[e] = (_Float16)(hh ? hi : lo);
        }
        v8f c0 = {}; v8f c1 = {};
        c0 = wmma16(af, bf0, c0);
        c1 = wmma16(af, bf1, c1);
        // epilogue: sigmoid + bias, mask diagonal, running max/min over j
        #pragma unroll
        for (int r = 0; r < 8; r++) {
            int jj = jt * 16 + r + (hh ? 8 : 0);   // C row M
            bool diag = (jj == i);
            float g = sigf(c0[r] + bias0);
            emax0 = fmaxf(emax0, diag ? 0.0f : g);
            amin0 = fminf(amin0, diag ? 1.0f : g);
            g = sigf(c1[r] + bias1);
            emax1 = fmaxf(emax1, diag ? 0.0f : g);
            amin1 = fminf(amin1, diag ? 1.0f : g);
        }
    }
    // combine the two half-wave row groups
    emax0 = fmaxf(emax0, __shfl_xor(emax0, 16, 32));
    amin0 = fminf(amin0, __shfl_xor(amin0, 16, 32));
    emax1 = fmaxf(emax1, __shfl_xor(emax1, 16, 32));
    amin1 = fminf(amin1, __shfl_xor(amin1, 16, 32));
    if (lane < 16) {
        // reduce_pair layout: interleaved [e_d, a_d] over d = 0..31
        float* dst = red + (size_t)(bb * NN + i) * 64;
        dst[2 * col + 0]        = emax0;
        dst[2 * col + 1]        = amin0;
        dst[2 * (col + 16) + 0] = emax1;
        dst[2 * (col + 16) + 1] = amin1;
    }
}

// ---------------------------------------------------------------------------
// h1: rows (B*n = 16384) x K=128 [g0|g1|red_g2] x N=32, swish(sigmoid(.)) out.
// One wave per 16-row M-tile; 4 K-steps x 2 N-tiles of WMMA. Source dispatch
// keys on the unrolled kc (compile-time); in-block K offset folds into the
// load address.
// ---------------------------------------------------------------------------
__global__ __launch_bounds__(32) void k_h1(const float* __restrict__ g0,
                                           const float* __restrict__ g1,
                                           const float* __restrict__ red,
                                           const float* __restrict__ W2_1,
                                           const float* __restrict__ b2_1,
                                           float* __restrict__ h1) {
    int tile = blockIdx.x;         // 0..1023
    int lane = threadIdx.x;
    int col = lane & 15;
    int hh8 = (lane >> 4) << 3;    // 0 or 8
    int Ra = tile * 16 + (lane & 15);   // A-fragment row
    int bba = Ra >> 6, ia = Ra & 63;
    const float* g1row  = g1  + (size_t)(bba * NN + ia) * DD;
    const float* redrow = red + (size_t)(bba * NN + ia) * 64;
    const float* g0row  = g0  + (size_t)bba * DD;

    v8f c0 = {}, c1 = {};
    #pragma unroll
    for (int kc = 0; kc < 4; kc++) {
        v16h af, bf0, bf1;
        #pragma unroll
        for (int e = 0; e < 16; e++) {
            const int kb = frag_kb(e);     // compile-time
            int kk = kb + hh8;             // runtime 0..31, address math only
            float a;
            if (kc == 0)      a = g0row[kk];
            else if (kc == 1) a = g1row[kk];
            else              a = redrow[(kc - 2) * 32 + kk];
            af[e] = (_Float16)a;
            int k = kc * 32 + kk;
            bf0[e] = (_Float16)W2_1[col * 128 + k];
            bf1[e] = (_Float16)W2_1[(col + 16) * 128 + k];
        }
        c0 = wmma16(af, bf0, c0);
        c1 = wmma16(af, bf1, c1);
    }
    float bias0 = b2_1[col], bias1 = b2_1[col + 16];
    #pragma unroll
    for (int r = 0; r < 8; r++) {
        int R = tile * 16 + r + hh8;
        float s = sigf(c0[r] + bias0);
        h1[(size_t)R * DD + col] = swishf(s);             // swish(sigmoid(z))
        s = sigf(c1[r] + bias1);
        h1[(size_t)R * DD + col + 16] = swishf(s);
    }
}

// ---------------------------------------------------------------------------
// conv1d over t (kernel 3, pad 1) as WMMA: M = t (16 rows), K = kt*32+ci (96),
// N = co (32). Fused bias + swish, writes final (b,t,n,D) output.
// Boundary rows: unconditional load from a clamped (valid) row + cndmask zero.
// ---------------------------------------------------------------------------
__global__ __launch_bounds__(32) void k_conv(const float* __restrict__ h1,
                                             const float* __restrict__ Wc,
                                             const float* __restrict__ bc,
                                             float* __restrict__ out) {
    int blk = blockIdx.x;          // 0..1023: b*64 + n
    int b = blk >> 6, nidx = blk & 63;
    int lane = threadIdx.x;
    int col = lane & 15;
    int hh8 = (lane >> 4) << 3;    // 0 or 8
    int tA = lane & 15;            // A-fragment row = t

    // hint the h1 rows for this (b, n) into cache (global_prefetch_b8)
    __builtin_prefetch(h1 + (size_t)((b * TT + tA) * NN + nidx) * DD, 0, 3);

    v8f c0 = {}, c1 = {};
    #pragma unroll
    for (int kt = 0; kt < 3; kt++) {
        int ts = tA + kt - 1;
        bool inb = (ts >= 0) && (ts < TT);
        int tclamp = inb ? ts : 0;     // always a valid row
        const float* src = h1 + (size_t)((b * TT + tclamp) * NN + nidx) * DD;
        v16h af, bf0, bf1;
        #pragma unroll
        for (int e = 0; e < 16; e++) {
            const int kb = frag_kb(e);
            int ci = kb + hh8;                             // 0..31
            float tmp = src[ci];
            af[e]  = (_Float16)(inb ? tmp : 0.0f);
            bf0[e] = (_Float16)Wc[col * (DD * 3) + ci * 3 + kt];
            bf1[e] = (_Float16)Wc[(col + 16) * (DD * 3) + ci * 3 + kt];
        }
        c0 = wmma16(af, bf0, c0);
        c1 = wmma16(af, bf1, c1);
    }
    float bias0 = bc[col], bias1 = bc[col + 16];
    #pragma unroll
    for (int r = 0; r < 8; r++) {
        int t = r + hh8;
        size_t base = (size_t)((b * TT + t) * NN + nidx) * DD;
        out[base + col]      = swishf(c0[r] + bias0);
        out[base + col + 16] = swishf(c1[r] + bias1);
    }
}

extern "C" void kernel_launch(void* const* d_in, const int* in_sizes, int n_in,
                              void* d_out, int out_size, void* d_ws, size_t ws_size,
                              hipStream_t stream) {
    const float* x    = (const float*)d_in[0];   // (256,64,2)
    const float* W1_0 = (const float*)d_in[1];   // (32,4)
    const float* b1_0 = (const float*)d_in[2];   // (32,)
    const float* W1_1 = (const float*)d_in[3];   // (32,8)
    const float* b1_1 = (const float*)d_in[4];   // (32,)
    const float* W1_2 = (const float*)d_in[5];   // (32,10)
    const float* b1_2 = (const float*)d_in[6];   // (32,)
    const float* W2_1 = (const float*)d_in[7];   // (32,128)
    const float* b2_1 = (const float*)d_in[8];   // (32,)
    const float* Wc   = (const float*)d_in[9];   // (32,32,3)
    const float* bc   = (const float*)d_in[10];  // (32,)
    float* out = (float*)d_out;                  // (16,16,64,32)

    // workspace carving (floats): g0 | g1 | red_g2 | h1  (~8.03 MB total)
    float* g0  = (float*)d_ws;                       // 256*32
    float* g1  = g0 + BB * DD;                       // 256*64*32
    float* red = g1 + (size_t)BB * NN * DD;          // 256*64*64
    float* h1  = red + (size_t)BB * NN * 64;         // 256*64*32

    k_g0<<<BB, 64, 0, stream>>>(x, W1_0, b1_0, g0);
    k_g1<<<(BB * NN + 255) / 256, 256, 0, stream>>>(x, W1_1, b1_1, g1);
    k_g2red<<<BB * NN, 32, 0, stream>>>(x, W1_2, b1_2, red);
    k_h1<<<(BB * NN) / 16, 32, 0, stream>>>(g0, g1, red, W2_1, b2_1, h1);
    k_conv<<<Bb * NN, 32, 0, stream>>>(h1, Wc, bc, out);
}
